// MultiHeadAtt_61959198212242
// MI455X (gfx1250) — compile-verified, matrix operand order checked by reading, not run on hardware
//
#include <hip/hip_runtime.h>
#include <hip/hip_bf16.h>
#include <stdint.h>

typedef _Float16 v16h __attribute__((ext_vector_type(16)));
typedef _Float16 v8h  __attribute__((ext_vector_type(8)));
typedef float    v8f  __attribute__((ext_vector_type(8)));
typedef int      v4i  __attribute__((ext_vector_type(4)));

constexpr int Bc  = 4;
constexpr int Sc  = 2048;
constexpr int Dc  = 1024;
constexpr int Hc  = 16;
constexpr int HDc = 64;
constexpr int OUTc = 256;

// ---- CDNA5 async global->LDS path (ASYNCcnt), guarded for robustness ------
#if defined(__has_builtin)
#if __has_builtin(__builtin_amdgcn_global_load_async_to_lds_b128)
#define HAVE_ASYNC_LDS 1
#else
#define HAVE_ASYNC_LDS 0
#endif
#if __has_builtin(__builtin_amdgcn_s_wait_asynccnt)
#define HAVE_WAIT_ASYNC 1
#else
#define HAVE_WAIT_ASYNC 0
#endif
#else
#define HAVE_ASYNC_LDS 0
#define HAVE_WAIT_ASYNC 0
#endif

typedef __attribute__((address_space(1))) v4i gv4i;   // global int4
typedef __attribute__((address_space(3))) v4i lv4i;   // LDS int4

// 16 bytes per lane: global -> LDS (async DMA when available).
// Casts go through uintptr_t (inttoptr) to sidestep addrspace cast rules;
// low 32 bits of a flat LDS pointer are the LDS offset per the ISA aperture
// mapping, so the AS3 truncation is value-correct.
__device__ inline void async_cp16(void* lds, const void* gsrc) {
#if HAVE_ASYNC_LDS
    gv4i* g = (gv4i*)(unsigned long long)(uintptr_t)gsrc;
    lv4i* l = (lv4i*)(unsigned int)(uintptr_t)lds;
    __builtin_amdgcn_global_load_async_to_lds_b128(g, l, 0, 0);
#else
    *(v8h*)lds = *(const v8h*)gsrc;
#endif
}

__device__ inline void wait_async0() {
#if HAVE_ASYNC_LDS
#if HAVE_WAIT_ASYNC
    __builtin_amdgcn_s_wait_asynccnt(0);
#else
    asm volatile("s_wait_asynccnt 0x0" ::: "memory");
#endif
#endif
}

__device__ inline v8f zero8() {
    v8f z;
#pragma unroll
    for (int i = 0; i < 8; ++i) z[i] = 0.0f;
    return z;
}

// Load a 16x32 f16 A/B fragment row for this lane from an LDS row pointer.
// Layout per ISA 7.12.2: lanes 0-15 hold K = 0..7,16..23 ; lanes 16-31 hold
// K = 8..15,24..31. grp = lane>>4. Two contiguous 16-byte chunks.
__device__ inline v16h load_frag(const _Float16* rowPtr, int grp) {
    const v8h lo = *(const v8h*)(rowPtr + grp * 8);
    const v8h hi = *(const v8h*)(rowPtr + grp * 8 + 16);
    v16h r;
#pragma unroll
    for (int i = 0; i < 8; ++i) { r[i] = lo[i]; r[i + 8] = hi[i]; }
    return r;
}

__device__ inline v8f wmma32(v16h a, v16h b, v8f c) {
    return __builtin_amdgcn_wmma_f32_16x16x32_f16(
        /*neg_a=*/false, a, /*neg_b=*/false, b,
        /*c_mod=*/(short)0, c, /*reuse_a=*/false, /*reuse_b=*/false);
}

// ---------------------------------------------------------------------------
// Kernel 1: QKV projection.  C[M=8192, N=1024] = X[M,K=1024] * W[K,N] + bias
// Output written as f16 in [B, H, S, HD] layout (head-transposed).
// Register-pipelined: tile k+1 global loads overlap tile k WMMAs.
// ---------------------------------------------------------------------------
__global__ void __launch_bounds__(256)
proj_kernel(const float* __restrict__ X, const float* __restrict__ W,
            const float* __restrict__ bias, _Float16* __restrict__ out)
{
    constexpr int BM = 128, BN = 128, BK = 32, LDS_S = 40; // 40 halves = 80B stride
    constexpr int Kdim = 1024, Nglob = 1024, NT = Kdim / BK;
    __shared__ __align__(16) _Float16 As[BM * LDS_S];
    __shared__ __align__(16) _Float16 Bs[BN * LDS_S];

    const int tid  = threadIdx.x;
    const int lane = tid & 31, wid = tid >> 5;
    const int waveM = wid & 3, waveN = wid >> 2;
    const int l16 = lane & 15, grp = lane >> 4;
    const int m0 = blockIdx.x * BM, n0 = blockIdx.y * BN;

    v8f acc[2][4];
#pragma unroll
    for (int i = 0; i < 2; ++i)
#pragma unroll
        for (int j = 0; j < 4; ++j) acc[i][j] = zero8();

    float4 aReg[4], bReg[4];
    auto load_tiles = [&](int k0) {
#pragma unroll
        for (int c = 0; c < 4; ++c) {
            const int idx = tid + c * 256;
            const int row = idx >> 3, c4 = idx & 7;
            aReg[c] = *(const float4*)(X + (size_t)(m0 + row) * Kdim + k0 + c4 * 4);
        }
#pragma unroll
        for (int c = 0; c < 4; ++c) {
            const int idx = tid + c * 256;
            const int kk = idx >> 5, n4 = idx & 31;
            bReg[c] = *(const float4*)(W + (size_t)(k0 + kk) * Nglob + n0 + n4 * 4);
        }
    };
    auto store_tiles = [&]() {
#pragma unroll
        for (int c = 0; c < 4; ++c) {
            const int idx = tid + c * 256;
            const int row = idx >> 3, c4 = idx & 7;
            _Float16* d = &As[row * LDS_S + c4 * 4];
            d[0] = (_Float16)aReg[c].x; d[1] = (_Float16)aReg[c].y;
            d[2] = (_Float16)aReg[c].z; d[3] = (_Float16)aReg[c].w;
        }
#pragma unroll
        for (int c = 0; c < 4; ++c) {
            const int idx = tid + c * 256;
            const int kk = idx >> 5, n4 = idx & 31;
            Bs[(n4 * 4 + 0) * LDS_S + kk] = (_Float16)bReg[c].x;
            Bs[(n4 * 4 + 1) * LDS_S + kk] = (_Float16)bReg[c].y;
            Bs[(n4 * 4 + 2) * LDS_S + kk] = (_Float16)bReg[c].z;
            Bs[(n4 * 4 + 3) * LDS_S + kk] = (_Float16)bReg[c].w;
        }
    };

    load_tiles(0);
    for (int kt = 0; kt < NT; ++kt) {
        store_tiles();
        __syncthreads();
        if (kt + 1 < NT) load_tiles((kt + 1) * BK);  // overlap with WMMAs below

        v16h afr[2], bfr[4];
#pragma unroll
        for (int im = 0; im < 2; ++im)
            afr[im] = load_frag(&As[(waveM * 32 + im * 16 + l16) * LDS_S], grp);
#pragma unroll
        for (int in = 0; in < 4; ++in)
            bfr[in] = load_frag(&Bs[(waveN * 64 + in * 16 + l16) * LDS_S], grp);
#pragma unroll
        for (int im = 0; im < 2; ++im)
#pragma unroll
            for (int in = 0; in < 4; ++in)
                acc[im][in] = wmma32(afr[im], bfr[in], acc[im][in]);
        __syncthreads();
    }

    // Epilogue: bias add, convert f16, scatter to [B,H,S,HD]
#pragma unroll
    for (int im = 0; im < 2; ++im)
#pragma unroll
        for (int in = 0; in < 4; ++in)
#pragma unroll
            for (int r = 0; r < 8; ++r) {
                const int mLoc = waveM * 32 + im * 16 + grp * 8 + r;
                const int nLoc = waveN * 64 + in * 16 + l16;
                const int gm = m0 + mLoc;   // b*S + s
                const int gn = n0 + nLoc;   // h*HD + hd
                const float v = acc[im][in][r] + bias[gn];
                const int b = gm >> 11, s = gm & 2047;
                const int h = gn >> 6, hd = gn & 63;
                out[(((size_t)(b * Hc + h) * Sc + s) * HDc) + hd] = (_Float16)v;
            }
}

// ---------------------------------------------------------------------------
// Kernel 2: fused attention per (b,h): energy = Q K^T / 8, softmax_1, P*V.
// 128 queries/block (8 waves, 16 rows each). Double-buffered 64-key tiles:
// K tiles stream via async global->LDS DMA (ASYNCcnt) while WMMAs run on the
// other buffer; V is transposed through VGPRs. Denominator = 1 + sum(exp),
// exactly the reference softmax_1 (energies are small; no max needed).
// ---------------------------------------------------------------------------
__global__ void __launch_bounds__(256)
attn_kernel(const _Float16* __restrict__ Q, const _Float16* __restrict__ K,
            const _Float16* __restrict__ V, _Float16* __restrict__ X)
{
    constexpr int LQ = 72;  // 72 halves = 144B row stride (16B aligned)
    constexpr int NT = Sc / 64;
    __shared__ __align__(16) _Float16 QP[128 * LQ];      // Q tile, then P tile
    __shared__ __align__(16) _Float16 Ks[2 * 64 * LQ];   // [buf][key][hd]
    __shared__ __align__(16) _Float16 Vt[2 * 64 * LQ];   // [buf][hd][key] (V^T)

    const int tid  = threadIdx.x;
    const int lane = tid & 31, wid = tid >> 5;
    const int l16 = lane & 15, grp = lane >> 4;
    const int bh = blockIdx.y;            // b*H + h
    const int b  = bh >> 4, h = bh & 15;
    const int q0 = blockIdx.x * 128;

    const _Float16* Qp = Q + (size_t)bh * Sc * HDc;
    const _Float16* Kp = K + (size_t)bh * Sc * HDc;
    const _Float16* Vp = V + (size_t)bh * Sc * HDc;

    auto stage_kv = [&](int kb, int buf) {
        _Float16* KsB = &Ks[buf * 64 * LQ];
#pragma unroll
        for (int c = 0; c < 2; ++c) {
            const int idx = tid + c * 256;
            const int row = idx >> 3, c8 = idx & 7;
            async_cp16(&KsB[row * LQ + c8 * 8],
                       Kp + (size_t)(kb + row) * HDc + c8 * 8);
        }
#pragma unroll
        for (int c = 0; c < 2; ++c) {
            const int idx = tid + c * 256;
            const int row = idx >> 3, c8 = idx & 7;
            const v8h hv = *(const v8h*)(Vp + (size_t)(kb + row) * HDc + c8 * 8);
#pragma unroll
            for (int j = 0; j < 8; ++j)
                Vt[buf * 64 * LQ + (c8 * 8 + j) * LQ + row] = hv[j];
        }
    };

    // Stage Q tile (async) + first K/V tile, then pull Q fragments to regs.
#pragma unroll
    for (int c = 0; c < 4; ++c) {
        const int idx = tid + c * 256;
        const int row = idx >> 3, c8 = idx & 7;
        async_cp16(&QP[row * LQ + c8 * 8],
                   Qp + (size_t)(q0 + row) * HDc + c8 * 8);
    }
    stage_kv(0, 0);
    wait_async0();
    __syncthreads();

    v16h qfr[2];
#pragma unroll
    for (int ks = 0; ks < 2; ++ks)
        qfr[ks] = load_frag(&QP[(wid * 16 + l16) * LQ] + ks * 32, grp);
    __syncthreads();  // everyone done reading QP; it becomes the P buffer

    v8f oacc[4];
#pragma unroll
    for (int i = 0; i < 4; ++i) oacc[i] = zero8();
    float dpart[8];
#pragma unroll
    for (int r = 0; r < 8; ++r) dpart[r] = 0.0f;

    for (int kt = 0; kt < NT; ++kt) {
        const int cur = kt & 1;
        wait_async0();       // this wave's pending tile (kt) has landed
        __syncthreads();     // all waves' tile kt visible; buf cur^1 is free
        if (kt + 1 < NT) stage_kv((kt + 1) * 64, cur ^ 1);  // DMA overlaps WMMA

        const _Float16* KsB = &Ks[cur * 64 * LQ];
        const _Float16* VtB = &Vt[cur * 64 * LQ];

        // energy = Q K^T  (M=16 queries, N=64 keys -> 4 n-subtiles, K=HD=64)
        v8f e[4];
#pragma unroll
        for (int in = 0; in < 4; ++in) {
            e[in] = zero8();
#pragma unroll
            for (int ks = 0; ks < 2; ++ks) {
                const v16h bfr = load_frag(&KsB[(in * 16 + l16) * LQ] + ks * 32, grp);
                e[in] = wmma32(qfr[ks], bfr, e[in]);
            }
        }
        // exp, partial denominator, stage P (f16) into this wave's QP rows
#pragma unroll
        for (int in = 0; in < 4; ++in)
#pragma unroll
            for (int r = 0; r < 8; ++r) {
                const float ev = __expf(e[in][r] * 0.125f);
                dpart[r] += ev;
                const int prow = wid * 16 + grp * 8 + r;
                const int pcol = in * 16 + l16;
                QP[prow * LQ + pcol] = (_Float16)ev;
            }

        // O += P * V  (M=16 queries, K=64 keys -> 2 k-steps, N=64 hd)
#pragma unroll
        for (int in = 0; in < 4; ++in)
#pragma unroll
            for (int ks = 0; ks < 2; ++ks) {
                const v16h afr = load_frag(&QP[(wid * 16 + l16) * LQ] + ks * 32, grp);
                const v16h bfr = load_frag(&VtB[(in * 16 + l16) * LQ] + ks * 32, grp);
                oacc[in] = wmma32(afr, bfr, oacc[in]);
            }
    }

    // Reduce denominator across the 16 lanes holding each row (xor<16 stays
    // within each half-wave, matching the C-fragment row split).
#pragma unroll
    for (int r = 0; r < 8; ++r) {
#pragma unroll
        for (int m = 1; m < 16; m <<= 1)
            dpart[r] += __shfl_xor(dpart[r], m);
        dpart[r] = 1.0f / (1.0f + dpart[r]);   // softmax_1 denominator
    }

    // Normalize and write X in [B, S, D] f16 layout
#pragma unroll
    for (int in = 0; in < 4; ++in)
#pragma unroll
        for (int r = 0; r < 8; ++r) {
            const float v = oacc[in][r] * dpart[r];
            const int qrow = q0 + wid * 16 + grp * 8 + r;
            const int col  = h * HDc + in * 16 + l16;
            X[((size_t)b * Sc + qrow) * Dc + col] = (_Float16)v;
        }
}

// ---------------------------------------------------------------------------
// Kernel 3: final projection.  out[M=8192, N=256] = Xh[M,K=1024] * Wf + bf
// Same tiling as proj; A already f16; register-pipelined staging.
// ---------------------------------------------------------------------------
__global__ void __launch_bounds__(256)
final_kernel(const _Float16* __restrict__ Xh, const float* __restrict__ W,
             const float* __restrict__ bias, float* __restrict__ out)
{
    constexpr int BM = 128, BN = 128, BK = 32, LDS_S = 40;
    constexpr int Kdim = 1024, Nglob = 256, NT = Kdim / BK;
    __shared__ __align__(16) _Float16 As[BM * LDS_S];
    __shared__ __align__(16) _Float16 Bs[BN * LDS_S];

    const int tid  = threadIdx.x;
    const int lane = tid & 31, wid = tid >> 5;
    const int waveM = wid & 3, waveN = wid >> 2;
    const int l16 = lane & 15, grp = lane >> 4;
    const int m0 = blockIdx.x * BM, n0 = blockIdx.y * BN;

    v8f acc[2][4];
#pragma unroll
    for (int i = 0; i < 2; ++i)
#pragma unroll
        for (int j = 0; j < 4; ++j) acc[i][j] = zero8();

    v8h aReg[2];
    float4 bReg[4];
    auto load_tiles = [&](int k0) {
#pragma unroll
        for (int c = 0; c < 2; ++c) {
            const int idx = tid + c * 256;
            const int row = idx >> 2, c8 = idx & 3;
            aReg[c] = *(const v8h*)(Xh + (size_t)(m0 + row) * Kdim + k0 + c8 * 8);
        }
#pragma unroll
        for (int c = 0; c < 4; ++c) {
            const int idx = tid + c * 256;
            const int kk = idx >> 5, n4 = idx & 31;
            bReg[c] = *(const float4*)(W + (size_t)(k0 + kk) * Nglob + n0 + n4 * 4);
        }
    };
    auto store_tiles = [&]() {
#pragma unroll
        for (int c = 0; c < 2; ++c) {
            const int idx = tid + c * 256;
            const int row = idx >> 2, c8 = idx & 3;
            *(v8h*)(&As[row * LDS_S + c8 * 8]) = aReg[c];
        }
#pragma unroll
        for (int c = 0; c < 4; ++c) {
            const int idx = tid + c * 256;
            const int kk = idx >> 5, n4 = idx & 31;
            Bs[(n4 * 4 + 0) * LDS_S + kk] = (_Float16)bReg[c].x;
            Bs[(n4 * 4 + 1) * LDS_S + kk] = (_Float16)bReg[c].y;
            Bs[(n4 * 4 + 2) * LDS_S + kk] = (_Float16)bReg[c].z;
            Bs[(n4 * 4 + 3) * LDS_S + kk] = (_Float16)bReg[c].w;
        }
    };

    load_tiles(0);
    for (int kt = 0; kt < NT; ++kt) {
        store_tiles();
        __syncthreads();
        if (kt + 1 < NT) load_tiles((kt + 1) * BK);

        v16h afr[2], bfr[4];
#pragma unroll
        for (int im = 0; im < 2; ++im)
            afr[im] = load_frag(&As[(waveM * 32 + im * 16 + l16) * LDS_S], grp);
#pragma unroll
        for (int in = 0; in < 4; ++in)
            bfr[in] = load_frag(&Bs[(waveN * 64 + in * 16 + l16) * LDS_S], grp);
#pragma unroll
        for (int im = 0; im < 2; ++im)
#pragma unroll
            for (int in = 0; in < 4; ++in)
                acc[im][in] = wmma32(afr[im], bfr[in], acc[im][in]);
        __syncthreads();
    }

#pragma unroll
    for (int im = 0; im < 2; ++im)
#pragma unroll
        for (int in = 0; in < 4; ++in)
#pragma unroll
            for (int r = 0; r < 8; ++r) {
                const int mLoc = waveM * 32 + im * 16 + grp * 8 + r;
                const int nLoc = waveN * 64 + in * 16 + l16;
                const int gm = m0 + mLoc;
                const int gn = n0 + nLoc;
                out[(size_t)gm * OUTc + gn] = acc[im][in][r] + bias[gn];
            }
}

// ---------------------------------------------------------------------------
extern "C" void kernel_launch(void* const* d_in, const int* in_sizes, int n_in,
                              void* d_out, int out_size, void* d_ws, size_t ws_size,
                              hipStream_t stream)
{
    (void)in_sizes; (void)n_in; (void)out_size; (void)ws_size;
    const float* query = (const float*)d_in[0];
    const float* key   = (const float*)d_in[1];
    const float* value = (const float*)d_in[2];
    const float* Wq = (const float*)d_in[3];
    const float* bq = (const float*)d_in[4];
    const float* Wk = (const float*)d_in[5];
    const float* bk = (const float*)d_in[6];
    const float* Wv = (const float*)d_in[7];
    const float* bv = (const float*)d_in[8];
    const float* Wf = (const float*)d_in[9];
    const float* bf = (const float*)d_in[10];
    float* out = (float*)d_out;

    // Workspace layout (f16): Qh | Kh | Vh | Xh, each B*H*S*HD = 8,388,608 elems
    _Float16* ws = (_Float16*)d_ws;
    const size_t per = (size_t)Bc * Hc * Sc * HDc;
    _Float16* Qh = ws;
    _Float16* Kh = ws + per;
    _Float16* Vh = ws + 2 * per;
    _Float16* Xh = ws + 3 * per;

    const dim3 blk(256);
    const dim3 gproj(Bc * Sc / 128, Dc / 128);       // (64, 8)
    proj_kernel<<<gproj, blk, 0, stream>>>(query, Wq, bq, Qh);
    proj_kernel<<<gproj, blk, 0, stream>>>(key,   Wk, bk, Kh);
    proj_kernel<<<gproj, blk, 0, stream>>>(value, Wv, bv, Vh);

    attn_kernel<<<dim3(Sc / 128, Bc * Hc), blk, 0, stream>>>(Qh, Kh, Vh, Xh);

    final_kernel<<<dim3(Bc * Sc / 128, OUTc / 128), blk, 0, stream>>>(Xh, Wf, bf, out);
}
